// FullVITS_3942779977806
// MI455X (gfx1250) — matrix operand order repaired and checked by reading, not compiled, not run on hardware
//
#include <hip/hip_runtime.h>
#include <math.h>

typedef __attribute__((ext_vector_type(16))) _Float16 v16h;
typedef __attribute__((ext_vector_type(8)))  _Float16 v8h;
typedef __attribute__((ext_vector_type(2)))  _Float16 v2h;
typedef __attribute__((ext_vector_type(8)))  float    v8f;

#define BB   32
#define TT   512
#define DD   256
#define HH   4
#define NLAY 6
#define TOUT 2048
#define DURCAP 8

// Fragment-order LDS row stride (halves): 40 halves = 80B, keeps every
// 16-half fragment 16B-aligned and spreads banks.
#define FSTR 40

// Load a 16-half WMMA fragment stored contiguously in LDS (2x ds_load_b128).
__device__ inline v16h ldfrag(const _Float16* p)
{
    v8h lo = *(const v8h*)p;
    v8h hi = *(const v8h*)(p + 8);
    return __builtin_shufflevector(lo, hi, 0, 1, 2, 3, 4, 5, 6, 7,
                                   8, 9, 10, 11, 12, 13, 14, 15);
}

// ---------------------------------------------------------------------------
// Generic WMMA GEMM with implicit im2col (conv1d) addressing.
//   C[m,n] = act( sum_k A_im2col[m,k] * W[k,n] + bias[n] + Res[m,n] )
// k = ks*CI + ci. A source row = m + ks - pad, valid iff 0 <= t+ks-pad < Tlen.
//
// REQUIREMENTS (all call sites satisfy):
//  * Tlen power of two; CI, K multiples of 32; CI > 32 when KS > 1
//  * M and N multiples of 64 (no epilogue bounds checks)
//  * when KS > 1, A points into the interior of the workspace with at least
//    pad*CI floats of addressable slack on both sides (loads are issued
//    unconditionally and masked with a value-select, not control flow)
//
// wmode 0: W row-major [K,N] (x@W).  wmode 1: conv layout W[n][ci][ks].
// Tile 64x64, BK=32, 128 threads (4 waves); wave = 16 rows x 64 cols.
// LDS tiles stored pre-swizzled into WMMA fragment order:
//   A: As2[row*FSTR + (k>>4)*16 + (k&15)]
//   B: Bs2[col*FSTR + (k>>4)*16 + (k&15)]   (transposed, fragment-contiguous)
// All staging addresses are maintained incrementally: no mul/div/branches in
// the K loop, only pointer bumps.
// ---------------------------------------------------------------------------
__global__ __launch_bounds__(128)
void wmma_gemm(const float* __restrict__ A, const float* __restrict__ W,
               const float* __restrict__ bias, const float* __restrict__ Res,
               float* __restrict__ C, int M, int N, int K,
               int CI, int KS, int pad, int Tlen, int relu, int wmode)
{
    __shared__ _Float16 As2[64 * FSTR];
    __shared__ _Float16 Bs2[64 * FSTR];
    const int tid  = threadIdx.x;
    const int lane = tid & 31, wv = tid >> 5;
    const int hi   = lane >> 4, l15 = lane & 15;
    const int m0 = blockIdx.y * 64, n0 = blockIdx.x * 64;

    // ---- K-invariant staging state -----------------------------------------
    // A staging: this thread covers k-offset cA (fixed) on rows rA0 + 8i.
    const int cA  = (tid & 15) * 2;
    const int rA0 = tid >> 4;
    const int hcA  = cA >> 4;
    const int posA = cA & 15;
    const float* pA[8];
    int  ttA[8];
#pragma unroll
    for (int i = 0; i < 8; ++i) {
        int m  = m0 + rA0 + i * 8;
        int mc = (m < M) ? m : (M - 1);            // clamp: keep address in-bounds
        int t  = mc & (Tlen - 1);                  // Tlen is a power of two
        ttA[i] = (m < M) ? (t - pad) : (int)0xc0000000;  // poison -> always invalid
        pA[i]  = A + ((long long)(mc - pad) * CI + cA);
    }
    // B staging: this thread covers column cnB (fixed), k-offsets ckB0 + 4i.
    const int cnB  = tid & 63;
    const int ckB0 = (tid >> 6) * 2;
    const bool nvB = (n0 + cnB) < N;
    const int  ncB = nvB ? (n0 + cnB) : (N - 1);   // clamped address column
    const int wstride = (wmode == 0) ? N : KS;     // pair-element stride
    // unified W base pointer + per-unroll stride (both layouts are affine)
    const float* wptr = (wmode == 0)
        ? (W + (size_t)ckB0 * N + ncB)
        : (W + (long long)ncB * CI * KS + (long long)ckB0 * KS);
    const long long wistep  = (wmode == 0) ? (long long)4 * N : (long long)4 * KS;
    const long long wadv    = (wmode == 0) ? (long long)32 * N : (long long)32 * KS;
    const long long wadvWrp = (wmode == 0) ? (long long)32 * N
                                           : ((long long)(32 - CI) * KS + 1);

    v8f acc[4];
#pragma unroll
    for (int nt = 0; nt < 4; ++nt)
#pragma unroll
        for (int i = 0; i < 8; ++i) acc[nt][i] = 0.0f;

    int ks = 0, cb = 0;               // k0 = ks*CI + cb, maintained incrementally
    for (int k0 = 0; k0 < K; k0 += 32) {
        // ---- phase 1: issue ALL staging loads (pipelined, no waits between)
        float2 fa[8];
#pragma unroll
        for (int i = 0; i < 8; ++i)
            fa[i] = *(const float2*)pA[i];
        float2 fw[8];
#pragma unroll
        for (int i = 0; i < 8; ++i) {
            const float* src = wptr + i * wistep;
            fw[i].x = src[0]; fw[i].y = src[wstride];
        }
        // prefetch next K-slab (speculative, gfx1250 global_prefetch path)
        if (k0 + 32 < K) {
            __builtin_prefetch(pA[0] + 32, 0, 0);
            if (wmode == 0)
                __builtin_prefetch(wptr + wadv, 0, 0);
        }
        // ---- phase 2: select + convert + LDS store ------------------------
#pragma unroll
        for (int i = 0; i < 8; ++i) {
            int st = ttA[i] + ks;
            bool ok = (st >= 0) && (st < Tlen);
            v2h pr;
            pr[0] = (_Float16)(ok ? fa[i].x : 0.0f);
            pr[1] = (_Float16)(ok ? fa[i].y : 0.0f);
            *(v2h*)&As2[(rA0 + i * 8) * FSTR + hcA * 16 + posA] = pr;
        }
#pragma unroll
        for (int i = 0; i < 8; ++i) {
            int ck = ckB0 + i * 4;
            v2h pr;
            pr[0] = (_Float16)(nvB ? fw[i].x : 0.0f);
            pr[1] = (_Float16)(nvB ? fw[i].y : 0.0f);
            *(v2h*)&Bs2[cnB * FSTR + (ck >> 4) * 16 + (ck & 15)] = pr;
        }
        // advance incremental staging state for next slab
#pragma unroll
        for (int i = 0; i < 8; ++i) pA[i] += 32;
        cb += 32;
        if (cb >= CI) { cb = 0; ++ks; wptr += wadvWrp; }
        else          { wptr += wadv; }
        __syncthreads();

        // ---- fragments: batch all LDS reads, then back-to-back WMMAs -------
        v16h a = ldfrag(&As2[(wv * 16 + l15) * FSTR + hi * 16]);
        v16h bfr[4];
#pragma unroll
        for (int nt = 0; nt < 4; ++nt)
            bfr[nt] = ldfrag(&Bs2[(nt * 16 + l15) * FSTR + hi * 16]);
#pragma unroll
        for (int nt = 0; nt < 4; ++nt)
            acc[nt] = __builtin_amdgcn_wmma_f32_16x16x32_f16(
                false, a, false, bfr[nt], (short)0, acc[nt], false, false);
        __syncthreads();
    }

    // ---- epilogue: D layout — VGPR r -> row r + hi*8, col = l15 ------------
    // M, N are multiples of 64 at every call site: no bounds checks needed.
#pragma unroll
    for (int nt = 0; nt < 4; ++nt)
#pragma unroll
        for (int r = 0; r < 8; ++r) {
            int row = m0 + wv * 16 + r + hi * 8;
            int col = n0 + nt * 16 + l15;
            float v = acc[nt][r];
            if (bias) v += bias[col];
            if (Res)  v += Res[(size_t)row * N + col];
            if (relu) v = fmaxf(v, 0.0f);
            C[(size_t)row * N + col] = v;
        }
}

// ---------------------------------------------------------------------------
// WMMA attention: one wave per (b, h, 16-query tile). dh=64, T=512.
// QKV layout: [B*T, 3*D]; q at col h*64, k at 256+h*64, v at 512+h*64.
// ---------------------------------------------------------------------------
__global__ __launch_bounds__(32)
void attn_kernel(const float* __restrict__ QKV, float* __restrict__ O)
{
    const int lane = threadIdx.x;
    const int qt = blockIdx.x, h = blockIdx.y, bb = blockIdx.z;
    const int hi = lane >> 4, l15 = lane & 15;
    const int RS = 3 * DD;
    const size_t base = (size_t)bb * TT * RS;

    __shared__ float S[16][TT + 8];

    // ---- S = Q K^T * scale -------------------------------------------------
    for (int kt = 0; kt < TT / 16; ++kt) {
        v8f acc;
#pragma unroll
        for (int i = 0; i < 8; ++i) acc[i] = 0.0f;
#pragma unroll
        for (int kc = 0; kc < 2; ++kc) {
            const float2* q2 = (const float2*)(QKV + base + (size_t)(qt * 16 + l15) * RS
                                               + h * 64 + kc * 32);
            const float2* k2 = (const float2*)(QKV + base + (size_t)(kt * 16 + l15) * RS
                                               + DD + h * 64 + kc * 32);
            // batch loads, then convert
            float2 qa[8], kb[8];
#pragma unroll
            for (int v = 0; v < 8; ++v) {
                int ia = (((v & 4) ? 16 : 0) + hi * 8 + 2 * (v & 3)) >> 1;
                qa[v] = q2[ia];
                kb[v] = k2[ia];
            }
            v16h a, bf;
#pragma unroll
            for (int v = 0; v < 8; ++v) {
                a[2 * v]      = (_Float16)qa[v].x;
                a[2 * v + 1]  = (_Float16)qa[v].y;
                bf[2 * v]     = (_Float16)kb[v].x;
                bf[2 * v + 1] = (_Float16)kb[v].y;
            }
            acc = __builtin_amdgcn_wmma_f32_16x16x32_f16(
                false, a, false, bf, (short)0, acc, false, false);
        }
#pragma unroll
        for (int r = 0; r < 8; ++r)
            S[r + hi * 8][kt * 16 + l15] = acc[r] * 0.125f;   // 1/sqrt(64)
    }
    __syncthreads();

    // ---- softmax rows (lanes 0..15, one row each) --------------------------
    if (lane < 16) {
        float mx = -1e30f;
        for (int j = 0; j < TT; ++j) mx = fmaxf(mx, S[lane][j]);
        float sm = 0.0f;
        for (int j = 0; j < TT; ++j) { float e = expf(S[lane][j] - mx); S[lane][j] = e; sm += e; }
        float inv = 1.0f / sm;
        for (int j = 0; j < TT; ++j) S[lane][j] *= inv;
    }
    __syncthreads();

    // ---- O = P V -----------------------------------------------------------
    v8f accO[4];
#pragma unroll
    for (int nt = 0; nt < 4; ++nt)
#pragma unroll
        for (int i = 0; i < 8; ++i) accO[nt][i] = 0.0f;

    for (int kc = 0; kc < TT / 32; ++kc) {
        const float2* s2 = (const float2*)&S[l15][kc * 32];
        float2 sv[8];
#pragma unroll
        for (int v = 0; v < 8; ++v)
            sv[v] = s2[(((v & 4) ? 16 : 0) + hi * 8 + 2 * (v & 3)) >> 1];
        v16h a;
#pragma unroll
        for (int v = 0; v < 8; ++v) {
            a[2 * v]     = (_Float16)sv[v].x;
            a[2 * v + 1] = (_Float16)sv[v].y;
        }
#pragma unroll
        for (int nt = 0; nt < 4; ++nt) {
            // batch the 16 V-element loads, then convert
            float v0[8], v1[8];
#pragma unroll
            for (int v = 0; v < 8; ++v) {
                int kb = ((v & 4) ? 16 : 0) + hi * 8 + 2 * (v & 3);          // key index within chunk
                const float* vrow = QKV + base + (size_t)(kc * 32 + kb) * RS
                                    + 2 * DD + h * 64 + nt * 16 + l15;
                v0[v] = vrow[0];
                v1[v] = vrow[RS];
            }
            v16h bf;
#pragma unroll
            for (int v = 0; v < 8; ++v) {
                bf[2 * v]     = (_Float16)v0[v];
                bf[2 * v + 1] = (_Float16)v1[v];
            }
            accO[nt] = __builtin_amdgcn_wmma_f32_16x16x32_f16(
                false, a, false, bf, (short)0, accO[nt], false, false);
        }
    }
#pragma unroll
    for (int nt = 0; nt < 4; ++nt)
#pragma unroll
        for (int r = 0; r < 8; ++r) {
            int row = qt * 16 + r + hi * 8;
            int col = h * 64 + nt * 16 + l15;
            O[(size_t)(bb * TT + row) * DD + col] = accO[nt][r];
        }
}

// ---------------------------------------------------------------------------
// Elementwise / small kernels
// ---------------------------------------------------------------------------
__global__ void embed_kernel(const int* __restrict__ text, const float* __restrict__ emb,
                             float* __restrict__ X, int n)
{
    int i = blockIdx.x * blockDim.x + threadIdx.x;
    if (i >= n) return;
    int d  = i & (DD - 1);
    int bt = i / DD;
    int t  = bt & (TT - 1);
    int tok = text[bt];
    int j  = d >> 1;
    float freq = expf((float)(2 * j) * (-9.210340371976184f / (float)DD));
    float ang  = (float)t * freq;
    float pe   = (d & 1) ? cosf(ang) : sinf(ang);
    X[i] = emb[(size_t)tok * DD + d] + pe;
}

__global__ __launch_bounds__(256)
void layernorm_kernel(const float* __restrict__ Xin, float* __restrict__ Y,
                      const float* __restrict__ g, const float* __restrict__ b)
{
    __shared__ float red[256];
    int row = blockIdx.x, tid = threadIdx.x;
    float x = Xin[(size_t)row * DD + tid];
    red[tid] = x; __syncthreads();
    for (int s = 128; s > 0; s >>= 1) { if (tid < s) red[tid] += red[tid + s]; __syncthreads(); }
    float mean = red[0] * (1.0f / DD);
    __syncthreads();
    float d = x - mean;
    red[tid] = d * d; __syncthreads();
    for (int s = 128; s > 0; s >>= 1) { if (tid < s) red[tid] += red[tid + s]; __syncthreads(); }
    float var = red[0] * (1.0f / DD);
    Y[(size_t)row * DD + tid] = d * rsqrtf(var + 1e-5f) * g[tid] + b[tid];
}

__global__ __launch_bounds__(256)
void emotion_kernel(const float* __restrict__ emo, const float* __restrict__ We1,
                    const float* __restrict__ be1, const float* __restrict__ We2,
                    const float* __restrict__ be2, float* __restrict__ E2)
{
    __shared__ float e1[64];
    int b = blockIdx.x, tid = threadIdx.x;
    if (tid < 64) e1[tid] = fmaxf(emo[b] * We1[tid] + be1[tid], 0.0f);
    __syncthreads();
    float acc = be2[tid];
    for (int j = 0; j < 64; ++j) acc += e1[j] * We2[j * DD + tid];
    E2[b * DD + tid] = acc;
}

__global__ void fuse_kernel(const float* __restrict__ X, const float* __restrict__ E2,
                            float* __restrict__ F, int n)
{
    int i = blockIdx.x * blockDim.x + threadIdx.x;
    if (i >= n) return;
    int d = i & (DD - 1);
    int b = i / (DD * TT);
    F[i] = X[i] + E2[b * DD + d];
}

__global__ void bn_kernel(float* __restrict__ H, const float* __restrict__ g,
                          const float* __restrict__ bt, const float* __restrict__ m,
                          const float* __restrict__ v, int n)
{
    int i = blockIdx.x * blockDim.x + threadIdx.x;
    if (i >= n) return;
    int c = i & (DD - 1);
    H[i] = g[c] * (H[i] - m[c]) * rsqrtf(v[c] + 1e-5f) + bt[c];
}

__global__ void dur_kernel(const float* __restrict__ H, const float* __restrict__ wo,
                           const float* __restrict__ bo, int* __restrict__ dur, int n)
{
    int i = blockIdx.x * blockDim.x + threadIdx.x;
    if (i >= n) return;
    float ld = bo[0];
    for (int c = 0; c < DD; ++c) ld += H[(size_t)i * DD + c] * wo[c];
    float r = rintf(expf(ld));
    r = fminf(fmaxf(r, 1.0f), (float)DURCAP);
    dur[i] = (int)r;
}

__global__ __launch_bounds__(256)
void regulate_kernel(const int* __restrict__ dur, int* __restrict__ idx)
{
    __shared__ int pre[TT];
    __shared__ int sd[TT];
    int b = blockIdx.x, tid = threadIdx.x;
    for (int i = tid; i < TT; i += 256) sd[i] = dur[b * TT + i];
    __syncthreads();
    if (tid == 0) { int a = 0; for (int i = 0; i < TT; ++i) { pre[i] = a; a += sd[i]; } }
    __syncthreads();
    for (int t = tid; t < TOUT; t += 256) idx[b * TOUT + t] = TT;   // pad marker
    __syncthreads();
    for (int i = tid; i < TT; i += 256) {
        int s = pre[i], c = sd[i];
        for (int j = 0; j < c; ++j) { int p = s + j; if (p < TOUT) idx[b * TOUT + p] = i; }
    }
}

__global__ void gather_kernel(const float* __restrict__ F, const int* __restrict__ idx,
                              float* __restrict__ R, int n)
{
    int i = blockIdx.x * blockDim.x + threadIdx.x;
    if (i >= n) return;
    int d = i & (DD - 1);
    int t = (i / DD) & (TOUT - 1);
    int b = i / (DD * TOUT);
    int id = idx[b * TOUT + t];
    R[i] = (id >= TT) ? 0.0f : F[((size_t)b * TT + id) * DD + d];
}

__global__ void dec3_kernel(const float* __restrict__ Y, const float* __restrict__ w,
                            const float* __restrict__ bias, float* __restrict__ wav, int n)
{
    int i = blockIdx.x * blockDim.x + threadIdx.x;
    if (i >= n) return;
    int t = i & (TOUT - 1);
    int b = i / TOUT;
    float acc = bias[0];
    for (int ks = 0; ks < 5; ++ks) {
        int st = t + ks - 2;
        if (st < 0 || st >= TOUT) continue;
        const float* row = Y + ((size_t)b * TOUT + st) * 128;
        for (int c = 0; c < 128; ++c) acc += row[c] * w[c * 5 + ks];
    }
    wav[i] = acc;
}

// ---------------------------------------------------------------------------
// Launcher
// ---------------------------------------------------------------------------
extern "C" void kernel_launch(void* const* d_in, const int* in_sizes, int n_in,
                              void* d_out, int out_size, void* d_ws, size_t ws_size,
                              hipStream_t stream)
{
    const int*   text = (const int*)  d_in[0];
    const float* emo  = (const float*)d_in[1];
    const float* emb  = (const float*)d_in[2];
    const float* Wqkv = (const float*)d_in[3];  const float* bqkv = (const float*)d_in[4];
    const float* Wo   = (const float*)d_in[5];  const float* bo   = (const float*)d_in[6];
    const float* W1   = (const float*)d_in[7];  const float* b1   = (const float*)d_in[8];
    const float* W2   = (const float*)d_in[9];  const float* b2   = (const float*)d_in[10];
    const float* ln1g = (const float*)d_in[11]; const float* ln1b = (const float*)d_in[12];
    const float* ln2g = (const float*)d_in[13]; const float* ln2b = (const float*)d_in[14];
    const float* We1  = (const float*)d_in[15]; const float* be1  = (const float*)d_in[16];
    const float* We2  = (const float*)d_in[17]; const float* be2  = (const float*)d_in[18];
    const float* dpw1 = (const float*)d_in[19]; const float* dpb1 = (const float*)d_in[20];
    const float* dpg1 = (const float*)d_in[21]; const float* dpbt1= (const float*)d_in[22];
    const float* dpm1 = (const float*)d_in[23]; const float* dpv1 = (const float*)d_in[24];
    const float* dpw2 = (const float*)d_in[25]; const float* dpb2 = (const float*)d_in[26];
    const float* dpg2 = (const float*)d_in[27]; const float* dpbt2= (const float*)d_in[28];
    const float* dpm2 = (const float*)d_in[29]; const float* dpv2 = (const float*)d_in[30];
    const float* dpwo = (const float*)d_in[31]; const float* dpbo = (const float*)d_in[32];
    const float* dw1  = (const float*)d_in[33]; const float* db1  = (const float*)d_in[34];
    const float* dw2  = (const float*)d_in[35]; const float* db2  = (const float*)d_in[36];
    const float* dw3  = (const float*)d_in[37]; const float* db3  = (const float*)d_in[38];
    float* wav = (float*)d_out;

    // workspace layout (floats), with aliasing across phases
    const size_t SZ_X = (size_t)BB * TT * DD;           // 4.19 M
    float* ws    = (float*)d_ws;
    float* X     = ws;
    float* FUSED = X + SZ_X;
    float* RA    = FUSED + SZ_X;                        // 16.78 M : QKV -> REG
    float* RB    = RA + (size_t)BB * TOUT * DD;         //  8.39 M : ATT -> DEC2
    float* RC    = RB + (size_t)BB * TOUT * 128;        // 16.78 M : TMP/H1 -> DEC1
    float* RD    = RC + (size_t)BB * TOUT * DD;         // 16.78 M : FF1/H2
    float* E2    = RD + (size_t)BB * TT * 4 * DD;
    int*   DUR   = (int*)(E2 + BB * DD);
    int*   IDX   = DUR + BB * TT;

    const int M = BB * TT;                              // 16384
    auto gemm = [&](const float* A, const float* W, const float* bias, const float* Res,
                    float* C, int Mm, int N, int K, int CI, int KS, int pad, int Tlen,
                    int relu, int wmode) {
        dim3 g((N + 63) / 64, (Mm + 63) / 64);
        wmma_gemm<<<g, dim3(128), 0, stream>>>(A, W, bias, Res, C, Mm, N, K,
                                               CI, KS, pad, Tlen, relu, wmode);
    };

    // 1. embedding + positional encoding
    {
        int n = M * DD;
        embed_kernel<<<(n + 255) / 256, 256, 0, stream>>>(text, emb, X, n);
    }

    // 2. transformer encoder (post-norm, relu FFN)
    float* QKV = RA; float* ATT = RB; float* TMP = RC; float* FF1 = RD;
    for (int i = 0; i < NLAY; ++i) {
        gemm(X, Wqkv + (size_t)i * DD * 3 * DD, bqkv + (size_t)i * 3 * DD, nullptr,
             QKV, M, 3 * DD, DD, DD, 1, 0, 1, 0, 0);
        attn_kernel<<<dim3(TT / 16, HH, BB), dim3(32), 0, stream>>>(QKV, ATT);
        gemm(ATT, Wo + (size_t)i * DD * DD, bo + (size_t)i * DD, X,
             TMP, M, DD, DD, DD, 1, 0, 1, 0, 0);
        layernorm_kernel<<<M, 256, 0, stream>>>(TMP, X, ln1g + i * DD, ln1b + i * DD);
        gemm(X, W1 + (size_t)i * DD * 4 * DD, b1 + (size_t)i * 4 * DD, nullptr,
             FF1, M, 4 * DD, DD, DD, 1, 0, 1, 1, 0);
        gemm(FF1, W2 + (size_t)i * 4 * DD * DD, b2 + (size_t)i * DD, X,
             TMP, M, DD, 4 * DD, 4 * DD, 1, 0, 1, 0, 0);
        layernorm_kernel<<<M, 256, 0, stream>>>(TMP, X, ln2g + i * DD, ln2b + i * DD);
    }

    // 3. emotion fusion
    emotion_kernel<<<BB, 256, 0, stream>>>(emo, We1, be1, We2, be2, E2);
    {
        int n = M * DD;
        fuse_kernel<<<(n + 255) / 256, 256, 0, stream>>>(X, E2, FUSED, n);
    }

    // 4. duration predictor: conv(k3) + relu + BN, twice; then 1x1 -> durations
    float* H1 = RC; float* H2 = RD;
    gemm(FUSED, dpw1, dpb1, nullptr, H1, M, DD, 3 * DD, DD, 3, 1, TT, 1, 1);
    bn_kernel<<<(M * DD + 255) / 256, 256, 0, stream>>>(H1, dpg1, dpbt1, dpm1, dpv1, M * DD);
    gemm(H1, dpw2, dpb2, nullptr, H2, M, DD, 3 * DD, DD, 3, 1, TT, 1, 1);
    bn_kernel<<<(M * DD + 255) / 256, 256, 0, stream>>>(H2, dpg2, dpbt2, dpm2, dpv2, M * DD);
    dur_kernel<<<(M + 255) / 256, 256, 0, stream>>>(H2, dpwo, dpbo, DUR, M);

    // 5. length regulation -> REG [B, TOUT, D]
    regulate_kernel<<<BB, 256, 0, stream>>>(DUR, IDX);
    float* REG = RA;
    {
        int n = BB * TOUT * DD;
        gather_kernel<<<(n + 255) / 256, 256, 0, stream>>>(FUSED, IDX, REG, n);
    }

    // 6. decoder conv stack (k5 convs as WMMA implicit GEMMs)
    float* DEC1 = RC; float* DEC2 = RB;
    gemm(REG,  dw1, db1, nullptr, DEC1, BB * TOUT, 256, 256 * 5, 256, 5, 2, TOUT, 1, 1);
    gemm(DEC1, dw2, db2, nullptr, DEC2, BB * TOUT, 128, 256 * 5, 256, 5, 2, TOUT, 1, 1);
    dec3_kernel<<<(BB * TOUT + 255) / 256, 256, 0, stream>>>(DEC2, dw3, db3, wav, BB * TOUT);
}
